// MultiScaleAttention_36292473651411
// MI455X (gfx1250) — compile-verified, matrix operand order checked by reading, not run on hardware
//
#include <hip/hip_runtime.h>
#include <hip/hip_bf16.h>
#include <stdint.h>

#define B_SZ 16
#define N_TOK 1024
#define DIM 768
#define DOUT 768
#define NH 8
#define HD 96
#define LP 256            // pooled tokens: 16x16
#define LN_EPS 1e-5f

typedef __attribute__((ext_vector_type(16))) __bf16 bf16x16;
typedef __attribute__((ext_vector_type(8)))  float  floatx8;

union FragBF { bf16x16 f; uint4 q[2]; };
union AccU   { floatx8 f; float s[8]; };

__device__ __forceinline__ uint16_t f2bf(float x) {
  uint32_t u = __float_as_uint(x);
  uint32_t r = u + 0x7FFFu + ((u >> 16) & 1u);   // round-to-nearest-even
  return (uint16_t)(r >> 16);
}

// ---------------------------------------------------------------------------
// fp32 -> bf16 pre-conversion (vectorized x4)
// ---------------------------------------------------------------------------
__global__ __launch_bounds__(256) void cvt_bf16_kernel(
    const float* __restrict__ in, uint16_t* __restrict__ out, int n4)
{
  const int i = blockIdx.x * 256 + threadIdx.x;
  if (i < n4) {
    const float4 v = ((const float4*)in)[i];
    ushort4 o;
    o.x = f2bf(v.x); o.y = f2bf(v.y); o.z = f2bf(v.z); o.w = f2bf(v.w);
    ((ushort4*)out)[i] = o;
  }
}

// ---------------------------------------------------------------------------
// bf16 WMMA GEMM: C[M,N] = A[M,K] * B[K,N] (+ bias[N]); bf16 in, fp32 out.
// Block tile 128x128, 8 waves of 32x64, K in chunks of 32.
// Double-buffered LDS; A tile staged via async-to-LDS (ASYNCcnt),
// B tile register-prefetched then scattered transposed into LDS.
// ---------------------------------------------------------------------------
#define TM 128
#define TN 128
#define TK 32
#define LSTR 40   // padded LDS row stride (80 bytes, multiple of 16B)

__global__ __launch_bounds__(256) void gemm_bf16_wmma(
    const uint16_t* __restrict__ A, const uint16_t* __restrict__ Bm,
    const float* __restrict__ bias, float* __restrict__ C,
    int M, int N, int K)
{
  __shared__ __align__(16) uint16_t ldsA[2][TM * LSTR];
  __shared__ __align__(16) uint16_t ldsB[2][TN * LSTR];

  const int tid  = threadIdx.x;
  const int lane = tid & 31;
  const int wave = tid >> 5;
  const int wm = wave & 3;          // wave M index (x32)
  const int wn = wave >> 2;         // wave N index (x64)
  const int blockM = blockIdx.y * TM;
  const int blockN = blockIdx.x * TN;

  AccU acc[2][4];
#pragma unroll
  for (int i = 0; i < 2; ++i)
#pragma unroll
    for (int j = 0; j < 4; ++j)
#pragma unroll
      for (int r = 0; r < 8; ++r) acc[i][j].s[r] = 0.0f;

  // A async staging: 512 x 16B chunks; thread handles chunks tid and tid+256.
  const int ar0 = tid >> 2;               // rows 0..63
  const int ar1 = ar0 + 64;               // rows 64..127
  const int ak  = (tid & 3) * 8;          // 8-element (16B) column offset
  // B staging: thread loads 16 consecutive bf16 from one row of the 32x128 tile
  const int brow = tid >> 3;              // 0..31 (k within tile)
  const int bn   = (tid & 7) * 16;        // 0..112 (n within tile)

  // fragment addressing (16-bit A-matrix layout)
  const int lr = lane & 15;
  const int kb = (lane >> 4) * 8;

  auto issueA = [&](int buf, int k0) {
    const uint16_t* g0 = A + (size_t)(blockM + ar0) * K + k0 + ak;
    const uint16_t* g1 = A + (size_t)(blockM + ar1) * K + k0 + ak;
    uint32_t l0 = (uint32_t)(uintptr_t)&ldsA[buf][ar0 * LSTR + ak];
    uint32_t l1 = (uint32_t)(uintptr_t)&ldsA[buf][ar1 * LSTR + ak];
    asm volatile("global_load_async_to_lds_b128 %0, %1, off"
                 :: "v"(l0), "v"(g0) : "memory");
    asm volatile("global_load_async_to_lds_b128 %0, %1, off"
                 :: "v"(l1), "v"(g1) : "memory");
  };
  auto loadB = [&](int k0, uint4& b0, uint4& b1) {
    const uint16_t* gb = Bm + (size_t)(k0 + brow) * N + blockN + bn;
    b0 = *(const uint4*)gb;
    b1 = *(const uint4*)(gb + 8);
  };
  auto storeB = [&](int buf, const uint4& b0, const uint4& b1) {
    const uint16_t* s0 = (const uint16_t*)&b0;
    const uint16_t* s1 = (const uint16_t*)&b1;
#pragma unroll
    for (int i = 0; i < 8; ++i) ldsB[buf][(bn + i) * LSTR + brow] = s0[i];
#pragma unroll
    for (int i = 0; i < 8; ++i) ldsB[buf][(bn + 8 + i) * LSTR + brow] = s1[i];
  };
  auto compute = [&](int buf) {
    FragBF afr[2], bfr[4];
#pragma unroll
    for (int mi = 0; mi < 2; ++mi) {
      const uint16_t* p = &ldsA[buf][(wm * 32 + mi * 16 + lr) * LSTR + kb];
      afr[mi].q[0] = *(const uint4*)(p);
      afr[mi].q[1] = *(const uint4*)(p + 16);
    }
#pragma unroll
    for (int ni = 0; ni < 4; ++ni) {
      const uint16_t* p = &ldsB[buf][(wn * 64 + ni * 16 + lr) * LSTR + kb];
      bfr[ni].q[0] = *(const uint4*)(p);
      bfr[ni].q[1] = *(const uint4*)(p + 16);
    }
#pragma unroll
    for (int mi = 0; mi < 2; ++mi)
#pragma unroll
      for (int ni = 0; ni < 4; ++ni)
        acc[mi][ni].f = __builtin_amdgcn_wmma_f32_16x16x32_bf16(
            false, afr[mi].f, false, bfr[ni].f, (short)0, acc[mi][ni].f, false, false);
  };

  const int niter = K / TK;
  // prologue: stage tile 0 into buffer 0
  issueA(0, 0);
  {
    uint4 b0, b1;
    loadB(0, b0, b1);
    storeB(0, b0, b1);
  }
  asm volatile("s_wait_asynccnt 0" ::: "memory");
  __syncthreads();

  for (int it = 0; it < niter; ++it) {
    const int cur = it & 1;
    const int nxt = cur ^ 1;
    const bool pre = (it + 1 < niter);
    uint4 b0, b1;
    if (pre) {                       // issue next tile before computing current
      const int k0n = (it + 1) * TK;
      issueA(nxt, k0n);
      loadB(k0n, b0, b1);
    }
    compute(cur);
    if (pre) {
      storeB(nxt, b0, b1);
      asm volatile("s_wait_asynccnt 0" ::: "memory");
    }
    __syncthreads();
  }

  // epilogue: VGPR r holds M=r (lanes 0-15) / M=r+8 (lanes 16-31), N=lane%16
  const int rowAdd = (lane >> 4) * 8;
  const int lc = lane & 15;
#pragma unroll
  for (int mi = 0; mi < 2; ++mi)
#pragma unroll
    for (int ni = 0; ni < 4; ++ni) {
      const int row0 = blockM + wm * 32 + mi * 16 + rowAdd;
      const int col  = blockN + wn * 64 + ni * 16 + lc;
      const float bv = bias ? bias[col] : 0.0f;
#pragma unroll
      for (int r = 0; r < 8; ++r)
        C[(size_t)(row0 + r) * N + col] = acc[mi][ni].s[r] + bv;
    }
}

// ---------------------------------------------------------------------------
// Depthwise 3x3 stride-2 pool (pad 1) + LayerNorm over HD channels.
// Block: 96 threads (one channel each), one block per (pixel, b*h, q/k/v).
// ---------------------------------------------------------------------------
__global__ __launch_bounds__(96) void pool_ln_kernel(
    const float* __restrict__ qkv,
    const float* __restrict__ wq, const float* __restrict__ wk, const float* __restrict__ wv,
    const float* __restrict__ gq, const float* __restrict__ bq,
    const float* __restrict__ gk, const float* __restrict__ bk,
    const float* __restrict__ gv, const float* __restrict__ bv,
    float* __restrict__ outq, float* __restrict__ outk, float* __restrict__ outv)
{
  const int c   = threadIdx.x;   // 0..95
  const int pix = blockIdx.x;    // 0..255
  const int bh  = blockIdx.y;    // 0..127
  const int t   = blockIdx.z;    // 0:q 1:k 2:v
  const int b = bh >> 3, h = bh & 7;
  const int py = pix >> 4, px = pix & 15;

  const float* w; const float* g; const float* be; float* dst;
  if (t == 0)      { w = wq; g = gq; be = bq; dst = outq; }
  else if (t == 1) { w = wk; g = gk; be = bk; dst = outk; }
  else             { w = wv; g = gv; be = bv; dst = outv; }

  const float* base = qkv + (size_t)b * N_TOK * (3 * DOUT) + (size_t)t * DOUT + h * HD + c;

  float a = 0.0f;
#pragma unroll
  for (int kh = 0; kh < 3; ++kh) {
    const int iy = py * 2 - 1 + kh;
    if (iy < 0 || iy > 31) continue;
#pragma unroll
    for (int kw = 0; kw < 3; ++kw) {
      const int ix = px * 2 - 1 + kw;
      if (ix < 0 || ix > 31) continue;
      a += w[c * 9 + kh * 3 + kw] * base[(size_t)(iy * 32 + ix) * (3 * DOUT)];
    }
  }

  __shared__ float sbuf[HD];
  sbuf[c] = a;
  __syncthreads();
  float s = 0.0f, sq = 0.0f;
  for (int i = 0; i < HD; ++i) { const float v = sbuf[i]; s += v; sq += v * v; }
  const float mean = s * (1.0f / HD);
  const float var  = sq * (1.0f / HD) - mean * mean;
  const float nv = (a - mean) * rsqrtf(var + LN_EPS) * g[c] + be[c];
  dst[((size_t)bh * LP + pix) * HD + c] = nv;
}

// ---------------------------------------------------------------------------
// Attention: per (b,h), softmax(q*scale @ k^T) @ v + q.  256 tokens, HD=96.
// Block = 256 threads handles 64 q-rows; k then v staged in shared LDS region.
// Output written bf16 in [b, pix, h*HD + c] layout for the projection GEMM.
// ---------------------------------------------------------------------------
__global__ __launch_bounds__(256) void attn_kernel(
    const float* __restrict__ poolq, const float* __restrict__ poolk,
    const float* __restrict__ poolv, uint16_t* __restrict__ obuf)
{
  __shared__ float ldsP[64 * 256];      // softmax probabilities (64 rows)
  __shared__ float ldsQ[64 * HD];       // q rows (scores + residual)
  __shared__ float ldsKV[LP * HD];      // k, then v

  const int tid  = threadIdx.x;
  const int lane = tid & 31;
  const int wave = tid >> 5;
  const int bh = blockIdx.y;
  const int b = bh >> 3, h = bh & 7;
  const int rowBase = blockIdx.x * 64;

  const size_t bhOff = (size_t)bh * LP * HD;

  for (int i = tid; i < 64 * HD; i += 256) ldsQ[i]  = poolq[bhOff + (size_t)rowBase * HD + i];
  for (int i = tid; i < LP * HD; i += 256) ldsKV[i] = poolk[bhOff + i];
  __syncthreads();

  const float scale = 0.1020620726159658f;   // 96^-0.5

  for (int rr = 0; rr < 8; ++rr) {
    const int r = wave * 8 + rr;
    float s[8];
#pragma unroll
    for (int jj = 0; jj < 8; ++jj) s[jj] = 0.0f;
    for (int c = 0; c < HD; ++c) {
      const float qc = ldsQ[r * HD + c];
#pragma unroll
      for (int jj = 0; jj < 8; ++jj)
        s[jj] += qc * ldsKV[(lane + 32 * jj) * HD + c];
    }
    float mx = -3.0e38f;
#pragma unroll
    for (int jj = 0; jj < 8; ++jj) { s[jj] *= scale; mx = fmaxf(mx, s[jj]); }
#pragma unroll
    for (int off = 16; off >= 1; off >>= 1) mx = fmaxf(mx, __shfl_xor(mx, off, 32));
    float sum = 0.0f;
#pragma unroll
    for (int jj = 0; jj < 8; ++jj) { const float e = __expf(s[jj] - mx); s[jj] = e; sum += e; }
#pragma unroll
    for (int off = 16; off >= 1; off >>= 1) sum += __shfl_xor(sum, off, 32);
    const float inv = 1.0f / sum;
#pragma unroll
    for (int jj = 0; jj < 8; ++jj) ldsP[r * 256 + lane + 32 * jj] = s[jj] * inv;
  }
  __syncthreads();
  for (int i = tid; i < LP * HD; i += 256) ldsKV[i] = poolv[bhOff + i];  // v overwrites k
  __syncthreads();

  for (int rr = 0; rr < 8; ++rr) {
    const int r = wave * 8 + rr;
    float a0 = ldsQ[r * HD + lane];            // residual (+ pooled q)
    float a1 = ldsQ[r * HD + lane + 32];
    float a2 = ldsQ[r * HD + lane + 64];
    for (int j = 0; j < LP; ++j) {
      const float p = ldsP[r * 256 + j];
      a0 += p * ldsKV[j * HD + lane];
      a1 += p * ldsKV[j * HD + lane + 32];
      a2 += p * ldsKV[j * HD + lane + 64];
    }
    uint16_t* op = obuf + ((size_t)b * LP + rowBase + r) * DOUT + h * HD;
    op[lane] = f2bf(a0); op[lane + 32] = f2bf(a1); op[lane + 64] = f2bf(a2);
  }
}

// ---------------------------------------------------------------------------
extern "C" void kernel_launch(void* const* d_in, const int* in_sizes, int n_in,
                              void* d_out, int out_size, void* d_ws, size_t ws_size,
                              hipStream_t stream) {
  (void)in_sizes; (void)n_in; (void)out_size; (void)ws_size;
  const float* x      = (const float*)d_in[0];
  const float* w_qkv  = (const float*)d_in[1];
  const float* w_proj = (const float*)d_in[2];
  const float* b_proj = (const float*)d_in[3];
  const float* pw_q   = (const float*)d_in[4];
  const float* pw_k   = (const float*)d_in[5];
  const float* pw_v   = (const float*)d_in[6];
  const float* g_q = (const float*)d_in[7];  const float* bq = (const float*)d_in[8];
  const float* g_k = (const float*)d_in[9];  const float* bk = (const float*)d_in[10];
  const float* g_v = (const float*)d_in[11]; const float* bv = (const float*)d_in[12];

  const size_t nX    = (size_t)B_SZ * N_TOK * DIM;        // 12,582,912
  const size_t nWqkv = (size_t)DIM * 3 * DOUT;            //  1,769,472
  const size_t nWprj = (size_t)DOUT * DOUT;               //    589,824
  const size_t nQKV  = (size_t)B_SZ * N_TOK * 3 * DOUT;   // 37,748,736
  const size_t nPool = (size_t)B_SZ * NH * LP * HD;       //  3,145,728
  const size_t nO    = (size_t)B_SZ * LP * DOUT;          //  3,145,728

  char* wsb = (char*)d_ws;
  float*    qkv    = (float*)wsb;                 wsb += nQKV  * sizeof(float);
  float*    poolq  = (float*)wsb;                 wsb += nPool * sizeof(float);
  float*    poolk  = (float*)wsb;                 wsb += nPool * sizeof(float);
  float*    poolv  = (float*)wsb;                 wsb += nPool * sizeof(float);
  uint16_t* xbf    = (uint16_t*)wsb;              wsb += nX    * sizeof(uint16_t);
  uint16_t* wqkvbf = (uint16_t*)wsb;              wsb += nWqkv * sizeof(uint16_t);
  uint16_t* wprjbf = (uint16_t*)wsb;              wsb += nWprj * sizeof(uint16_t);
  uint16_t* obuf   = (uint16_t*)wsb;              wsb += nO    * sizeof(uint16_t);

  // 0) pre-convert operands to bf16
  cvt_bf16_kernel<<<(int)(nX / 4 / 256),    256, 0, stream>>>(x,      xbf,    (int)(nX / 4));
  cvt_bf16_kernel<<<(int)(nWqkv / 4 / 256), 256, 0, stream>>>(w_qkv,  wqkvbf, (int)(nWqkv / 4));
  cvt_bf16_kernel<<<(int)(nWprj / 4 / 256), 256, 0, stream>>>(w_proj, wprjbf, (int)(nWprj / 4));

  // 1) QKV GEMM: (16384 x 768) * (768 x 2304)
  dim3 g1((3 * DOUT) / TN, (B_SZ * N_TOK) / TM);
  gemm_bf16_wmma<<<g1, 256, 0, stream>>>(xbf, wqkvbf, nullptr, qkv,
                                         B_SZ * N_TOK, 3 * DOUT, DIM);
  // 2) depthwise pool + layernorm for q,k,v
  dim3 g2(LP, B_SZ * NH, 3);
  pool_ln_kernel<<<g2, 96, 0, stream>>>(qkv, pw_q, pw_k, pw_v,
                                        g_q, bq, g_k, bk, g_v, bv,
                                        poolq, poolk, poolv);
  // 3) attention + residual (writes bf16)
  dim3 g3(LP / 64, B_SZ * NH);
  attn_kernel<<<g3, 256, 0, stream>>>(poolq, poolk, poolv, obuf);
  // 4) projection GEMM: (4096 x 768) * (768 x 768) + bias
  dim3 g4(DOUT / TN, (B_SZ * LP) / TM);
  gemm_bf16_wmma<<<g4, 256, 0, stream>>>(obuf, wprjbf, b_proj, (float*)d_out,
                                         B_SZ * LP, DOUT, DOUT);
}
